// Net_55783035240677
// MI455X (gfx1250) — matrix operand order repaired and checked by reading, not compiled, not run on hardware
//
#include <hip/hip_runtime.h>
#include <hip/hip_bf16.h>
#include <math.h>

// ---------------------------------------------------------------------------
// GAT 2-layer forward for MI455X (gfx1250, wave32, WMMA).
// GEMMs via v_wmma_f32_16x16x32_bf16; edge softmax-aggregation via L2-resident
// atomic passes (all intermediates ~98MB fit in the 192MB L2).
// ---------------------------------------------------------------------------

typedef __attribute__((ext_vector_type(16))) __bf16 v16bf;
typedef __attribute__((ext_vector_type(8)))  float  v8f;

#define NEG_SLOPE 0.2f

__device__ __forceinline__ void atomicMaxF(float* addr, float val) {
  // Works for any sign given init = -inf: positive floats order as signed ints,
  // negative floats order inversely as unsigned ints.
  if (val >= 0.0f) atomicMax((int*)addr, __float_as_int(val));
  else             atomicMin((unsigned int*)addr, __float_as_uint(val));
}

// -------------------------- fill --------------------------------------------
__global__ void fill_kernel(float* __restrict__ p, float v, int n) {
  int i = blockIdx.x * blockDim.x + threadIdx.x;
  if (i < n) p[i] = v;
}

// -------------------------- GEMM1: h1 = x @ W1  ([N,256]x[256,64]) ----------
// 1 block = 16 nodes, 4 waves; wave w computes output cols [16w,16w+16).
__global__ void gemm1_wmma_kernel(const float* __restrict__ x,
                                  const float* __restrict__ W1,
                                  float* __restrict__ h1, int Nn) {
  const int node0 = blockIdx.x * 16;
  const int wave  = threadIdx.x >> 5;      // 0..3
  const int lane  = threadIdx.x & 31;
  const int m     = lane & 15;
  const int hi    = lane >> 4;
  int arow = node0 + m; if (arow >= Nn) arow = Nn - 1;   // clamp (ragged tile only)
  const float* __restrict__ xrow = x + (size_t)arow * 256;
  const int n = wave * 16 + m;             // B column / output column
  v8f acc = {};
  for (int kt = 0; kt < 256; kt += 32) {
    v16bf a, b;
#pragma unroll
    for (int r = 0; r < 8; ++r) {          // A 16x32 bf16 fragment layout
      const int k0 = kt + 2 * r + ((r >= 4) ? 8 : 0) + hi * 8;
      const float2 p = *(const float2*)(xrow + k0);
      a[2 * r]     = (__bf16)p.x;
      a[2 * r + 1] = (__bf16)p.y;
    }
#pragma unroll
    for (int r = 0; r < 8; ++r) {          // B 32x16 bf16 fragment layout
      const int k = kt + 2 * r + ((r >= 4) ? 8 : 0) + hi * 8;
      b[2 * r]     = (__bf16)W1[(size_t)k * 64 + n];
      b[2 * r + 1] = (__bf16)W1[(size_t)(k + 1) * 64 + n];
    }
    acc = __builtin_amdgcn_wmma_f32_16x16x32_bf16(false, a, false, b,
                                                  (short)0, acc, false, false);
  }
  // C/D layout: lane covers rows M = r + 8*hi (r=0..7), column N = lane&15.
  if (node0 + 16 <= Nn) {                  // uniform fast path: no EXEC churn
    float* __restrict__ base = h1 + (size_t)(node0 + hi * 8) * 64 + n;
#pragma unroll
    for (int r = 0; r < 8; ++r) base[(size_t)r * 64] = acc[r];
  } else {
#pragma unroll
    for (int r = 0; r < 8; ++r) {
      const int node = node0 + r + hi * 8;
      if (node < Nn) h1[(size_t)node * 64 + n] = acc[r];
    }
  }
}

// -------------------------- GEMM2: h2 = hact @ W2  ([N,64]x[64,40]) ---------
// 1 block = 16 nodes, 3 waves (48 cols, guarded to 40).
__global__ void gemm2_wmma_kernel(const float* __restrict__ hact,
                                  const float* __restrict__ W2,
                                  float* __restrict__ h2, int Nn) {
  const int node0 = blockIdx.x * 16;
  const int wave  = threadIdx.x >> 5;      // 0..2
  const int lane  = threadIdx.x & 31;
  const int m     = lane & 15;
  const int hi    = lane >> 4;
  int arow = node0 + m; if (arow >= Nn) arow = Nn - 1;
  const float* __restrict__ hrow = hact + (size_t)arow * 64;
  const int n = wave * 16 + m;             // 0..47
  v8f acc = {};
  for (int kt = 0; kt < 64; kt += 32) {
    v16bf a, b;
#pragma unroll
    for (int r = 0; r < 8; ++r) {
      const int k0 = kt + 2 * r + ((r >= 4) ? 8 : 0) + hi * 8;
      const float2 p = *(const float2*)(hrow + k0);
      a[2 * r]     = (__bf16)p.x;
      a[2 * r + 1] = (__bf16)p.y;
    }
#pragma unroll
    for (int r = 0; r < 8; ++r) {
      const int k = kt + 2 * r + ((r >= 4) ? 8 : 0) + hi * 8;
      const float w0 = (n < 40) ? W2[(size_t)k * 40 + n] : 0.0f;
      const float w1 = (n < 40) ? W2[(size_t)(k + 1) * 40 + n] : 0.0f;
      b[2 * r] = (__bf16)w0;  b[2 * r + 1] = (__bf16)w1;
    }
    acc = __builtin_amdgcn_wmma_f32_16x16x32_bf16(false, a, false, b,
                                                  (short)0, acc, false, false);
  }
  if (n < 40) {
    if (node0 + 16 <= Nn) {                // fast path
      float* __restrict__ base = h2 + (size_t)(node0 + hi * 8) * 40 + n;
#pragma unroll
      for (int r = 0; r < 8; ++r) base[(size_t)r * 40] = acc[r];
    } else {
#pragma unroll
      for (int r = 0; r < 8; ++r) {
        const int node = node0 + r + hi * 8;
        if (node < Nn) h2[(size_t)node * 40 + n] = acc[r];
      }
    }
  }
}

// -------------------------- attention scores per node -----------------------
template <int H, int D>
__global__ void scores_kernel(const float* __restrict__ h,
                              const float* __restrict__ asrc,
                              const float* __restrict__ adst,
                              float* __restrict__ ss, float* __restrict__ sd,
                              int Nn) {
  const int i = blockIdx.x * blockDim.x + threadIdx.x;   // i = n*H + head
  if (i >= Nn * H) return;
  const int n  = i / H;
  const int hh = i - n * H;
  const float* __restrict__ hp = h + ((size_t)n * H + hh) * D;
  float vs = 0.0f, vd = 0.0f;
#pragma unroll
  for (int d = 0; d < D; ++d) {
    const float v = hp[d];
    vs = fmaf(v, asrc[hh * D + d], vs);
    vd = fmaf(v, adst[hh * D + d], vd);
  }
  ss[i] = vs;  sd[i] = vd;
}

// -------------------------- edge passes -------------------------------------
template <int H>
__device__ __forceinline__ float edge_score(const int* __restrict__ ei, int E,
                                            const float* __restrict__ ss,
                                            const float* __restrict__ sd,
                                            int e, int hh, int& s, int& d) {
  s = ei[e];
  d = ei[E + e];
  float v = ss[s * H + hh] + sd[d * H + hh];
  return (v > 0.0f) ? v : NEG_SLOPE * v;     // leaky_relu
}

template <int H>
__global__ void edge_max_kernel(const int* __restrict__ ei, int E,
                                const float* __restrict__ ss,
                                const float* __restrict__ sd,
                                float* __restrict__ m) {
  const int i = blockIdx.x * blockDim.x + threadIdx.x;
  if (i >= E * H) return;
  const int e = i / H, hh = i - e * H;
  int s, d;
  const float v = edge_score<H>(ei, E, ss, sd, e, hh, s, d);
  atomicMaxF(&m[d * H + hh], v);
}

template <int H>
__global__ void edge_sum_kernel(const int* __restrict__ ei, int E,
                                const float* __restrict__ ss,
                                const float* __restrict__ sd,
                                const float* __restrict__ m,
                                float* __restrict__ z) {
  const int i = blockIdx.x * blockDim.x + threadIdx.x;
  if (i >= E * H) return;
  const int e = i / H, hh = i - e * H;
  int s, d;
  const float v = edge_score<H>(ei, E, ss, sd, e, hh, s, d);
  atomicAdd(&z[d * H + hh], __expf(v - m[d * H + hh]));
}

template <int H, int D>
__global__ void edge_aggr_kernel(const int* __restrict__ ei, int E,
                                 const float* __restrict__ ss,
                                 const float* __restrict__ sd,
                                 const float* __restrict__ m,
                                 const float* __restrict__ z,
                                 const float* __restrict__ hfeat,
                                 float* __restrict__ outa) {
  const int i = blockIdx.x * blockDim.x + threadIdx.x;
  if (i >= E * H) return;
  const int e = i / H, hh = i - e * H;
  int s, d;
  const float v = edge_score<H>(ei, E, ss, sd, e, hh, s, d);
  const float alpha = __expf(v - m[d * H + hh]) / (z[d * H + hh] + 1e-16f);
  const float* __restrict__ hp = hfeat + ((size_t)s * H + hh) * D;
  float* __restrict__ op = outa + ((size_t)d * H + hh) * D;
#pragma unroll
  for (int dd = 0; dd < D; ++dd) atomicAdd(&op[dd], hp[dd] * alpha);
}

// -------------------------- activations / output ----------------------------
template <int F>
__global__ void elu_bias_kernel(float* __restrict__ a, const float* __restrict__ b,
                                int total) {
  const int i = blockIdx.x * blockDim.x + threadIdx.x;
  if (i >= total) return;
  const float v = a[i] + b[i % F];
  a[i] = (v > 0.0f) ? v : (__expf(v) - 1.0f);
}

template <int C>
__global__ void logsoftmax_kernel(const float* __restrict__ acc,
                                  const float* __restrict__ b,
                                  float* __restrict__ out, int Nn) {
  const int n = blockIdx.x * blockDim.x + threadIdx.x;
  if (n >= Nn) return;
  const float* __restrict__ p = acc + (size_t)n * C;
  float mx = -INFINITY;
#pragma unroll
  for (int c = 0; c < C; ++c) mx = fmaxf(mx, p[c] + b[c]);
  float sum = 0.0f;
#pragma unroll
  for (int c = 0; c < C; ++c) sum += __expf(p[c] + b[c] - mx);
  const float lse = mx + __logf(sum);
  float* __restrict__ op = out + (size_t)n * C;
#pragma unroll
  for (int c = 0; c < C; ++c) op[c] = p[c] + b[c] - lse;
}

// ---------------------------------------------------------------------------
extern "C" void kernel_launch(void* const* d_in, const int* in_sizes, int n_in,
                              void* d_out, int out_size, void* d_ws, size_t ws_size,
                              hipStream_t stream) {
  const float* x   = (const float*)d_in[0];
  const int*   ei  = (const int*)  d_in[1];
  const float* W1  = (const float*)d_in[2];
  const float* a1s = (const float*)d_in[3];
  const float* a1d = (const float*)d_in[4];
  const float* b1  = (const float*)d_in[5];
  const float* W2  = (const float*)d_in[6];
  const float* a2s = (const float*)d_in[7];
  const float* a2d = (const float*)d_in[8];
  const float* b2  = (const float*)d_in[9];
  float* out = (float*)d_out;

  const int Nn = in_sizes[0] / 256;   // 100000 nodes
  const int E  = in_sizes[1] / 2;     // 1.6M edges
  const int H = 8, C = 40;

  // Workspace layout (floats). All intermediates stay L2-resident (~98 MB).
  float* ws = (float*)d_ws;
  size_t o = 0;
  float* h1   = ws + o; o += (size_t)Nn * 64;   // layer1 features
  float* s1s  = ws + o; o += (size_t)Nn * H;
  float* s1d  = ws + o; o += (size_t)Nn * H;
  float* m1   = ws + o; o += (size_t)Nn * H;
  float* z1   = ws + o; o += (size_t)Nn * H;
  float* acc1 = ws + o; o += (size_t)Nn * 64;   // layer1 out -> elu in place
  float* h2   = ws + o; o += (size_t)Nn * C;    // layer2 features
  float* s2s  = ws + o; o += (size_t)Nn;
  float* s2d  = ws + o; o += (size_t)Nn;
  float* m2   = ws + o; o += (size_t)Nn;
  float* z2   = ws + o; o += (size_t)Nn;
  float* acc2 = ws + o; o += (size_t)Nn * C;    // layer2 out
  (void)ws_size; (void)n_in; (void)out_size;

  const int T = 256;
  auto cdiv = [](long a, long b) { return (int)((a + b - 1) / b); };
  const int nodeTiles = cdiv(Nn, 16);

  // ---- layer 1 ----
  gemm1_wmma_kernel<<<nodeTiles, 128, 0, stream>>>(x, W1, h1, Nn);
  scores_kernel<8, 8><<<cdiv((long)Nn * H, T), T, 0, stream>>>(h1, a1s, a1d, s1s, s1d, Nn);
  fill_kernel<<<cdiv((long)Nn * H, T), T, 0, stream>>>(m1, -INFINITY, Nn * H);
  fill_kernel<<<cdiv((long)Nn * H, T), T, 0, stream>>>(z1, 0.0f, Nn * H);
  fill_kernel<<<cdiv((long)Nn * 64, T), T, 0, stream>>>(acc1, 0.0f, Nn * 64);
  edge_max_kernel<8><<<cdiv((long)E * H, T), T, 0, stream>>>(ei, E, s1s, s1d, m1);
  edge_sum_kernel<8><<<cdiv((long)E * H, T), T, 0, stream>>>(ei, E, s1s, s1d, m1, z1);
  edge_aggr_kernel<8, 8><<<cdiv((long)E * H, T), T, 0, stream>>>(ei, E, s1s, s1d, m1, z1, h1, acc1);
  elu_bias_kernel<64><<<cdiv((long)Nn * 64, T), T, 0, stream>>>(acc1, b1, Nn * 64);

  // ---- layer 2 ----
  gemm2_wmma_kernel<<<nodeTiles, 96, 0, stream>>>(acc1, W2, h2, Nn);
  scores_kernel<1, 40><<<cdiv((long)Nn, T), T, 0, stream>>>(h2, a2s, a2d, s2s, s2d, Nn);
  fill_kernel<<<cdiv((long)Nn, T), T, 0, stream>>>(m2, -INFINITY, Nn);
  fill_kernel<<<cdiv((long)Nn, T), T, 0, stream>>>(z2, 0.0f, Nn);
  fill_kernel<<<cdiv((long)Nn * C, T), T, 0, stream>>>(acc2, 0.0f, Nn * C);
  edge_max_kernel<1><<<cdiv((long)E, T), T, 0, stream>>>(ei, E, s2s, s2d, m2);
  edge_sum_kernel<1><<<cdiv((long)E, T), T, 0, stream>>>(ei, E, s2s, s2d, m2, z2);
  edge_aggr_kernel<1, 40><<<cdiv((long)E, T), T, 0, stream>>>(ei, E, s2s, s2d, m2, z2, h2, acc2);

  // ---- output: log_softmax(acc2 + b2) ----
  logsoftmax_kernel<40><<<cdiv((long)Nn, T), T, 0, stream>>>(acc2, b2, out, Nn);
}
